// ShidoTreeLSTMLayerTreeBase_13683765805745
// MI455X (gfx1250) — compile-verified, matrix operand order checked by reading, not run on hardware
//
#include <hip/hip_runtime.h>
#include <hip/hip_bf16.h>

// ---------------------------------------------------------------------------
// ShidoTreeLSTM layer for MI455X (gfx1250).  All GEMMs run on the WMMA pipe
// (v_wmma_f32_16x16x32_bf16, f32 accumulate); pointwise LSTM math in fp32.
// Tile staging uses GLOBAL_LOAD_ASYNC_TO_LDS_B128 (ASYNCcnt-tracked) when the
// toolchain exposes the builtin, else global_load+ds_store fallback.
// ---------------------------------------------------------------------------

typedef __bf16 bf16;
typedef __attribute__((ext_vector_type(8)))  __bf16 v8bf;
typedef __attribute__((ext_vector_type(16))) __bf16 v16bf;
typedef __attribute__((ext_vector_type(8)))  float  v8f;

#define NN   4096   // nodes
#define KCH  16     // max children (timesteps)
#define DD   256    // hidden dim
#define LDSROW 40   // padded bf16 elems per LDS row (80B: 16B aligned, conflict-free)

#if __has_builtin(__builtin_amdgcn_global_load_async_to_lds_b128) && \
    __has_builtin(__builtin_amdgcn_s_wait_asynccnt)
#define USE_ASYNC_LDS 1
// Builtin signature (per hipcc diagnostic): (int4vec AS1*, int4vec AS3*, Imm, Imm)
typedef int v4i_async __attribute__((vector_size(16)));
typedef __attribute__((address_space(1))) v4i_async g_v4i;
typedef __attribute__((address_space(3))) v4i_async l_v4i;
__device__ __forceinline__ void async_cp_b128(const void* gsrc, void* ldst) {
  __builtin_amdgcn_global_load_async_to_lds_b128(
      (g_v4i*)(void*)gsrc, (l_v4i*)ldst, 0, 0);
}
#else
#define USE_ASYNC_LDS 0
#endif

__device__ __forceinline__ bf16 f2bf(float x) {
  unsigned u = __float_as_uint(x);
  unsigned r = (u + 0x7FFFu + ((u >> 16) & 1u)) >> 16;   // RNE
  unsigned short h = (unsigned short)r;
  return __builtin_bit_cast(bf16, h);
}
__device__ __forceinline__ float sigm(float x) { return 1.0f / (1.0f + __expf(-x)); }

// Load a 16-element bf16 fragment row per the gfx1250 16-bit A/B layout:
// lane half (lane>>4) owns k = half*8..half*8+7 and 16+half*8..16+half*8+7.
__device__ __forceinline__ v16bf load_frag(const bf16* row, int half) {
  v8bf lo = *(const v8bf*)(row + half * 8);
  v8bf hi = *(const v8bf*)(row + 16 + half * 8);
  return __builtin_shufflevector(lo, hi, 0,1,2,3,4,5,6,7,8,9,10,11,12,13,14,15);
}

// ---------------------------------------------------------------------------
// C[M,Nc] = concat_k(A1[k<256], A2[k>=256]) @ Bt^T + bias
//   A1/A2: bf16, row-major with independent row strides (elements)
//   Bt:    bf16, [Nc, Kd] (i.e. B transposed) so B-fragments load contiguously
//   M % 128 == 0, Nc % 128 == 0, Kd % 32 == 0, Kd <= 512 (source switch at 256)
// Block: 256 threads (8 waves), tile 128x128; wave tile 32x64 (2x4 WMMA frags).
// ---------------------------------------------------------------------------
__global__ __launch_bounds__(256)
void gemm_bf16_wmma(const bf16* __restrict__ A1, long sA1,
                    const bf16* __restrict__ A2, long sA2,
                    const bf16* __restrict__ Bt,
                    const float* __restrict__ bias,
                    float* __restrict__ C,
                    int M, int Ncols, int Kd)
{
  __shared__ __align__(16) bf16 As[128 * LDSROW];
  __shared__ __align__(16) bf16 Bs[128 * LDSROW];
  const int tid  = threadIdx.x;
  const int bm0  = blockIdx.x * 128;
  const int bn0  = blockIdx.y * 128;
  const int wave = tid >> 5;
  const int lane = tid & 31;
  const int wm   = wave & 3;    // 4 waves along M
  const int wn   = wave >> 2;   // 2 waves along N
  const int lrow = lane & 15;
  const int half = lane >> 4;

  v8f acc[2][4] = {};

  for (int k0 = 0; k0 < Kd; k0 += 32) {
    const bf16* Asrc; long sA; int kofs;
    if (k0 < 256) { Asrc = A1; sA = sA1; kofs = k0; }
    else          { Asrc = A2; sA = sA2; kofs = k0 - 256; }

#pragma unroll
    for (int rep = 0; rep < 2; ++rep) {           // 512 x 16B units, 2 per thread
      int u = tid + rep * 256;
      int row = u >> 2, c = u & 3;
      const bf16* ga = Asrc + (long)(bm0 + row) * sA + (kofs + c * 8);
      const bf16* gb = Bt + (long)(bn0 + row) * (long)Kd + (k0 + c * 8);
      bf16* la = &As[row * LDSROW + c * 8];
      bf16* lb = &Bs[row * LDSROW + c * 8];
#if USE_ASYNC_LDS
      async_cp_b128(ga, la);
      async_cp_b128(gb, lb);
#else
      *(uint4*)la = *(const uint4*)ga;
      *(uint4*)lb = *(const uint4*)gb;
#endif
    }

    if (k0 + 32 < Kd) {                           // global_prefetch_b8 next tile
      int kn = k0 + 32;
      const bf16* P; long sP; int kof2;
      if (kn < 256) { P = A1; sP = sA1; kof2 = kn; }
      else          { P = A2; sP = sA2; kof2 = kn - 256; }
      int prow = tid >> 1, pc = tid & 1;
      __builtin_prefetch(P  + (long)(bm0 + prow) * sP + kof2 + pc * 16, 0, 1);
      __builtin_prefetch(Bt + (long)(bn0 + prow) * (long)Kd + kn + pc * 16, 0, 1);
    }
#if USE_ASYNC_LDS
    __builtin_amdgcn_s_wait_asynccnt(0);          // our async tile writes landed
#endif
    __syncthreads();

    v16bf af[2], bfv[4];
#pragma unroll
    for (int mf = 0; mf < 2; ++mf)
      af[mf] = load_frag(&As[(wm * 32 + mf * 16 + lrow) * LDSROW], half);
#pragma unroll
    for (int nf = 0; nf < 4; ++nf)
      bfv[nf] = load_frag(&Bs[(wn * 64 + nf * 16 + lrow) * LDSROW], half);

#pragma unroll
    for (int mf = 0; mf < 2; ++mf)
#pragma unroll
      for (int nf = 0; nf < 4; ++nf)
        acc[mf][nf] = __builtin_amdgcn_wmma_f32_16x16x32_bf16(
            false, af[mf], false, bfv[nf], (short)0, acc[mf][nf], false, false);
    __syncthreads();
  }

  // D layout: lane n = col0 + lane%16; VGPR j holds row row0 + j + 8*(lane/16)
#pragma unroll
  for (int mf = 0; mf < 2; ++mf)
#pragma unroll
    for (int nf = 0; nf < 4; ++nf) {
      int row0 = bm0 + wm * 32 + mf * 16;
      int col0 = bn0 + wn * 64 + nf * 16;
      int n = col0 + lrow;
      float badd = bias ? bias[n] : 0.0f;
#pragma unroll
      for (int j = 0; j < 8; ++j) {
        int m = row0 + j + 8 * half;
        C[(long)m * Ncols + n] = acc[mf][nf][j] + badd;
      }
    }
}

// ---------------------------------------------------------------------------
// Prep / gather / pointwise kernels
// ---------------------------------------------------------------------------
__global__ void len_kernel(const int* __restrict__ ind, int* __restrict__ len) {
  int n = blockIdx.x * blockDim.x + threadIdx.x;
  if (n < NN) {
    int c = 0;
#pragma unroll
    for (int k = 0; k < KCH; ++k) c += (ind[n * KCH + k] != -1);
    len[n] = c;
  }
}

// hg  = bf16 gathered child h          [N,K,D]
// hgr = bf16 reverse_sequence(hg, len) [N,K,D]
// cg  = fp32 gathered child c          [N,K,D]
__global__ void gather_kernel(const float* __restrict__ h_t, const float* __restrict__ c_t,
                              const int* __restrict__ ind, const int* __restrict__ len,
                              bf16* __restrict__ hg, bf16* __restrict__ hgr,
                              float* __restrict__ cg) {
  int n = blockIdx.x >> 4, k = blockIdx.x & 15, j = threadIdx.x;
  int idx  = ind[n * KCH + k];
  int safe = idx < 0 ? 0 : idx;
  long o = ((long)(n * KCH + k)) * DD + j;
  float hv = h_t[(long)safe * DD + j];
  hg[o] = f2bf(hv);
  cg[o] = c_t[(long)safe * DD + j];
  int L = len[n];
  int s = (k < L) ? (L - 1 - k) : k;
  int idx2  = ind[n * KCH + s];
  int safe2 = idx2 < 0 ? 0 : idx2;
  hgr[o] = f2bf(h_t[(long)safe2 * DD + j]);
}

__global__ void cvt_x_kernel(const float* __restrict__ x, bf16* __restrict__ xbf) {
  long i = (long)blockIdx.x * 256 + threadIdx.x;
  xbf[i] = f2bf(x[i]);
}

// W_w [256,1024] -> Wwt [1024,256] bf16
__global__ void wwt_kernel(const float* __restrict__ Ww, bf16* __restrict__ Wwt) {
  int nc = blockIdx.x, k = threadIdx.x;
  Wwt[(long)nc * 256 + k] = f2bf(Ww[(long)k * 1024 + nc]);
}

// [kernel;rec](g,d) [512,1024] -> Wcat[gd] [1024,512] bf16
__global__ void wcat_kernel(const float* __restrict__ kern, const float* __restrict__ rec,
                            bf16* __restrict__ Wcat) {
  int gd = blockIdx.x >> 10, nc = blockIdx.x & 1023;
#pragma unroll
  for (int r = 0; r < 2; ++r) {
    int k = threadIdx.x + r * 256;
    float v = (k < 256) ? kern[((long)gd * 256 + k) * 1024 + nc]
                        : rec [((long)gd * 256 + (k - 256)) * 1024 + nc];
    Wcat[((long)gd * 1024 + nc) * 512 + k] = f2bf(v);
  }
}

// fc_w[g] [512,256] -> fct[g] [256,512] bf16
__global__ void fct_kernel(const float* __restrict__ fcw, bf16* __restrict__ fct) {
  int g = blockIdx.x >> 8, nc = blockIdx.x & 255;
#pragma unroll
  for (int r = 0; r < 2; ++r) {
    int k = threadIdx.x + r * 256;
    fct[((long)g * 256 + nc) * 512 + k] = f2bf(fcw[((long)g * 512 + k) * 256 + nc]);
  }
}

__global__ void init_state_kernel(bf16* __restrict__ hbf, float* __restrict__ cst,
                                  const float* __restrict__ ih, const float* __restrict__ ic) {
  int n = blockIdx.x, j = threadIdx.x;
  hbf[(long)n * DD + j] = f2bf(ih[j]);
  cst[(long)n * DD + j] = ic[j];
}

// z [N,1024] gate order i,f,g,o (Keras).  Updates c (fp32) / h (bf16);
// streams h into fseq (f-gate, every t) or ylast (i/u/o, at t == len-1).
__global__ void lstm_pointwise(const float* __restrict__ z, float* __restrict__ cst,
                               bf16* __restrict__ hbf, const int* __restrict__ len,
                               int t, bf16* __restrict__ fseq, bf16* __restrict__ ylast) {
  int n = blockIdx.x, j = threadIdx.x;
  long zb = (long)n * 1024;
  float zi = z[zb + j];
  float zf = z[zb + 256 + j];
  float zg = z[zb + 512 + j];
  float zo = z[zb + 768 + j];
  long o = (long)n * DD + j;
  float c  = cst[o];
  float cn = sigm(zf) * c + sigm(zi) * tanhf(zg);
  float hn = sigm(zo) * tanhf(cn);
  cst[o] = cn;
  hbf[o] = f2bf(hn);
  if (fseq)  fseq[((long)(n * KCH + t)) * DD + j] = f2bf(hn);
  if (ylast && t == len[n] - 1) ylast[(long)n * 512 + j] = f2bf(hn);
}

// branch_f[n,j] = sum_t<L sigm(W_f_x + fseq_fc) * c_child * mask
__global__ void f_reduce_kernel(const float* __restrict__ Wx, const float* __restrict__ fso,
                                const float* __restrict__ cg, const int* __restrict__ len,
                                float* __restrict__ bf_out) {
  int n = blockIdx.x, j = threadIdx.x;
  float wf = Wx[(long)n * 1024 + j];       // f gate = first split of W_x
  int L = len[n];
  float acc = 0.0f;
  for (int t = 0; t < KCH; ++t)
    if (t < L)
      acc += sigm(wf + fso[((long)(n * KCH + t)) * DD + j]) *
             cg[((long)(n * KCH + t)) * DD + j];
  bf_out[(long)n * DD + j] = acc;
}

__global__ void finalize_kernel(const float* __restrict__ Wx,
                                const float* __restrict__ ti, const float* __restrict__ tu,
                                const float* __restrict__ to, const float* __restrict__ bfv,
                                float* __restrict__ out) {
  int n = blockIdx.x, j = threadIdx.x;
  long o = (long)n * DD + j;
  long wb = (long)n * 1024;
  float bi = sigm (ti[o] + Wx[wb + 256 + j]);
  float bu = tanhf(tu[o] + Wx[wb + 512 + j]);
  float bo = sigm (to[o] + Wx[wb + 768 + j]);
  float nc = bi * bu + bfv[o];
  float nh = bo * tanhf(nc);
  out[o] = nh;
  out[(long)NN * DD + o] = nc;
}

// ---------------------------------------------------------------------------
extern "C" void kernel_launch(void* const* d_in, const int* in_sizes, int n_in,
                              void* d_out, int out_size, void* d_ws, size_t ws_size,
                              hipStream_t stream) {
  const float* x         = (const float*)d_in[0];
  const float* h_tensor  = (const float*)d_in[1];
  const float* c_tensor  = (const float*)d_in[2];
  const int*   indice    = (const int*)  d_in[3];
  const float* W_w       = (const float*)d_in[4];
  const float* W_b       = (const float*)d_in[5];
  const float* lstm_kern = (const float*)d_in[6];
  const float* lstm_rec  = (const float*)d_in[7];
  const float* lstm_bias = (const float*)d_in[8];
  const float* init_h    = (const float*)d_in[9];
  const float* init_c    = (const float*)d_in[10];
  const float* fc_w      = (const float*)d_in[11];
  float* out = (float*)d_out;

  // ---- workspace carve (all 256B aligned) ----
  char* ws = (char*)d_ws;
  size_t off = 0;
  auto take = [&](size_t bytes) -> void* {
    void* p = ws + off;
    off += (bytes + 255) & ~(size_t)255;
    return p;
  };
  const size_t NKD = (size_t)NN * KCH * DD;
  bf16*  hgath   = (bf16*) take(NKD * 2);            // gathered child h (bf16)
  bf16*  hgathr  = (bf16*) take(NKD * 2);            // reversed sequence
  float* cgath   = (float*)take(NKD * 4);            // gathered child c (fp32)
  int*   lenb    = (int*)  take((size_t)NN * 4);
  float* Wx      = (float*)take((size_t)NN * 1024 * 4);
  bf16*  xbf     = (bf16*) take((size_t)NN * DD * 2);
  bf16*  Wwt     = (bf16*) take((size_t)1024 * 256 * 2);
  bf16*  Wcat    = (bf16*) take((size_t)8 * 1024 * 512 * 2);
  bf16*  fct     = (bf16*) take((size_t)4 * 256 * 512 * 2);
  bf16*  hstate  = (bf16*) take((size_t)NN * DD * 2);
  float* cstate  = (float*)take((size_t)NN * DD * 4);
  float* zbuf    = (float*)take((size_t)NN * 1024 * 4);
  bf16*  fseqf   = (bf16*) take(NKD * 2);
  bf16*  fseqb   = (bf16*) take(NKD * 2);
  bf16*  ylast   = (bf16*) take((size_t)3 * NN * 512 * 2);
  float* branchf = (float*)take((size_t)NN * DD * 4);
  float* tmp     = (float*)take((size_t)3 * NN * DD * 4);
  // f-seq fc output (67MB) aliases hgath+hgathr, which are dead by then.
  float* fseqout = (float*)hgath;

  // ---- prep ----
  len_kernel   <<<NN / 256, 256, 0, stream>>>(indice, lenb);
  gather_kernel<<<NN * KCH, 256, 0, stream>>>(h_tensor, c_tensor, indice, lenb,
                                              hgath, hgathr, cgath);
  cvt_x_kernel <<<NN,       256, 0, stream>>>(x, xbf);
  wwt_kernel   <<<1024,     256, 0, stream>>>(W_w, Wwt);
  wcat_kernel  <<<8 * 1024, 256, 0, stream>>>(lstm_kern, lstm_rec, Wcat);
  fct_kernel   <<<4 * 256,  256, 0, stream>>>(fc_w, fct);

  // ---- W_x = x @ W_w + W_b   [4096,1024] ----
  gemm_bf16_wmma<<<dim3(NN / 128, 1024 / 128), 256, 0, stream>>>(
      xbf, (long)DD, xbf, (long)DD, Wwt, W_b, Wx, NN, 1024, 256);

  // ---- 4 gates x 2 directions, T=16 scan ----
  for (int g = 0; g < 4; ++g) {
    for (int d = 0; d < 2; ++d) {
      int gd = g * 2 + d;
      init_state_kernel<<<NN, 256, 0, stream>>>(hstate, cstate,
                                                init_h + (long)gd * DD,
                                                init_c + (long)gd * DD);
      const bf16* seq = (d == 0) ? hgath : hgathr;
      for (int t = 0; t < KCH; ++t) {
        // z = [h_child_t | h_state] @ [kernel;rec]^T(g,d) + bias
        gemm_bf16_wmma<<<dim3(NN / 128, 1024 / 128), 256, 0, stream>>>(
            seq + (long)t * DD, (long)KCH * DD,   // A1: child-h at step t
            hstate, (long)DD,                     // A2: recurrent h
            Wcat + (size_t)gd * 1024 * 512,
            lstm_bias + (long)gd * 1024,
            zbuf, NN, 1024, 512);
        bf16* fdst = (g == 0) ? ((d == 0) ? fseqf : fseqb) : nullptr;
        bf16* ydst = (g > 0) ? (ylast + (size_t)(g - 1) * NN * 512 + (size_t)d * DD)
                             : nullptr;
        lstm_pointwise<<<NN, 256, 0, stream>>>(zbuf, cstate, hstate, lenb, t, fdst, ydst);
      }
    }
  }

  // ---- f branch: fseq_cat [65536,512] @ fc_w[0]^T -> [65536,256] ----
  gemm_bf16_wmma<<<dim3((NN * KCH) / 128, 256 / 128), 256, 0, stream>>>(
      fseqf, (long)DD, fseqb, (long)DD, fct, nullptr, fseqout, NN * KCH, 256, 512);
  f_reduce_kernel<<<NN, 256, 0, stream>>>(Wx, fseqout, cgath, lenb, branchf);

  // ---- i/u/o branches: y_last [4096,512] @ fc_w[g]^T ----
  for (int gi = 0; gi < 3; ++gi) {
    const bf16* yl = ylast + (size_t)gi * NN * 512;
    gemm_bf16_wmma<<<dim3(NN / 128, 256 / 128), 256, 0, stream>>>(
        yl, (long)512, yl + 256, (long)512,
        fct + (size_t)(gi + 1) * 256 * 512, nullptr,
        tmp + (size_t)gi * NN * DD, NN, 256, 512);
  }

  // ---- combine: new_c = i*u + f_branch; new_h = o*tanh(new_c) ----
  finalize_kernel<<<NN, 256, 0, stream>>>(Wx, tmp, tmp + (size_t)NN * DD,
                                          tmp + (size_t)2 * NN * DD, branchf, out);
}
